// EfficientBalancedAveragedHausdorffLoss_46213848105412
// MI455X (gfx1250) — compile-verified, matrix operand order checked by reading, not run on hardware
//
#include <hip/hip_runtime.h>

// ---------------------------------------------------------------------------
// Balanced averaged Hausdorff loss for MI455X (gfx1250, wave32, WMMA).
//
// dist^2(i,j) = |p_i|^2 + |t_j|^2 - 2 p_i . t_j
// Coordinates are integers <= 511 -> exact in f16; -2*coord exact in f16;
// all products/sums are integers < 2^24 -> f32 WMMA accumulation reproduces
// the reference dx^2+dy^2 bit-exactly.  |p|^2 is folded into the WMMA C
// operand; min-reduction happens in d^2 space (sqrt is monotone) so only 8
// sqrts per wave run after the target loop.
// ---------------------------------------------------------------------------

typedef __attribute__((ext_vector_type(16))) _Float16 v16h;
typedef __attribute__((ext_vector_type(8)))  float    v8f;

#define HH 512
#define WW 512
#define NSAMP 64            // 16 * 4
#define PIX (HH * WW)       // 262144
#define KCAP 1000           // per-sample point cap
#define LSTRIDE 1024        // padded list stride (points)
#define PTILES 63           // ceil(1000/16)
#define THRESH (0.3f + 1e-5f)
#define TSENT 20000.0f      // sentinel coord for padded targets (d^2 ~ 7.6e8)

struct __align__(8) F2 { float r, c; };

// ---------------------------------------------------------------------------
__global__ void habl_zero_kernel(float* out) {
    if (threadIdx.x == 0 && blockIdx.x == 0) out[0] = 0.0f;
}

// ---------------------------------------------------------------------------
// One block (1024 threads = 32 waves) per sample.  Each thread handles 4
// consecutive pixels via one b128 load per tensor.  Ordered stream
// compaction of 4 masks simultaneously: m1 (pred-only), t, m2 (target-only),
// p.  Row-major order is preserved: within a wave, earlier lanes' 4 pixels
// all precede this lane's; cross-wave prefix via a __shfl_up scan done by
// waves 0-3 (one mask each); running base across 64 chunk iterations.
// ---------------------------------------------------------------------------
__global__ void habl_extract_kernel(const float4* __restrict__ pred,
                                    const float4* __restrict__ target,
                                    F2* __restrict__ pts,
                                    int* __restrict__ counts) {
    const int s    = blockIdx.x;
    const int tid  = threadIdx.x;
    const int lane = tid & 31;
    const int wid  = tid >> 5;            // 0..31
    const float4* ps = pred   + (size_t)s * (PIX / 4);
    const float4* ts = target + (size_t)s * (PIX / 4);

    __shared__ int wpre[4][32];   // exclusive cross-wave prefix per mask
    __shared__ int wtot[4];       // per-chunk totals per mask
    __shared__ int base[4];       // running base (row-major order)
    __shared__ int wcnt[4][32];   // per-wave counts per mask
    if (tid < 4) base[tid] = 0;
    __syncthreads();

    const unsigned ltmask = (1u << lane) - 1u;

    for (int chunk = 0; chunk < PIX / 4; chunk += 1024) {
        const int q = chunk + tid;            // float4 index within sample
        const float4 pv = ps[q];
        const float4 tv = ts[q];
        const float pa[4] = {pv.x, pv.y, pv.z, pv.w};
        const float ta[4] = {tv.x, tv.y, tv.z, tv.w};

        bool bits[4][4];                       // [mask][pixel]
        #pragma unroll
        for (int k = 0; k < 4; ++k) {
            const bool pb = fabsf(pa[k] - 1.0f) <= THRESH;
            const bool tb = fabsf(ta[k] - 1.0f) <= THRESH;
            bits[0][k] = pb && !tb;   // m1: loss1 pred points
            bits[1][k] = tb;          // t : loss1 target points
            bits[2][k] = tb && !pb;   // m2: loss2 pred points
            bits[3][k] = pb;          // p : loss2 target points
        }

        int pre[4];      // pixels of earlier lanes in this wave
        #pragma unroll
        for (int l = 0; l < 4; ++l) {
            int p = 0, c = 0;
            #pragma unroll
            for (int k = 0; k < 4; ++k) {
                const unsigned bal = (unsigned)__ballot(bits[l][k]);
                p += __popc(bal & ltmask);
                c += __popc(bal);
            }
            pre[l] = p;
            if (lane == 0) wcnt[l][wid] = c;
        }
        __syncthreads();

        // waves 0-3: wave-parallel exclusive scan of the 32 wave-counts
        if (wid < 4) {
            const int l = wid;
            const int v = wcnt[l][lane];
            int inc = v;
            #pragma unroll
            for (int off = 1; off < 32; off <<= 1) {
                const int n = __shfl_up(inc, off, 32);
                if (lane >= off) inc += n;
            }
            wpre[l][lane] = inc - v;          // exclusive prefix
            if (lane == 31) wtot[l] = inc;    // chunk total
        }
        __syncthreads();

        const int pix0 = q * 4;               // first pixel of this thread
        #pragma unroll
        for (int l = 0; l < 4; ++l) {
            int slot = base[l] + wpre[l][wid] + pre[l];
            F2* dst = pts + (size_t)(s * 4 + l) * LSTRIDE;
            #pragma unroll
            for (int k = 0; k < 4; ++k) {
                if (bits[l][k]) {
                    if (slot < KCAP) {
                        const int idx = pix0 + k;
                        dst[slot].r = (float)(idx >> 9);   // W = 512
                        dst[slot].c = (float)(idx & 511);
                    }
                    ++slot;
                }
            }
        }
        __syncthreads();
        if (tid == 0) {
            #pragma unroll
            for (int l = 0; l < 4; ++l) base[l] += wtot[l];
        }
        // next iteration's first barrier orders the base update before reads
    }
    __syncthreads();
    if (tid < 4) counts[s * 4 + tid] = base[tid];   // uncapped totals
}

// ---------------------------------------------------------------------------
// One wave per (sample, loss, pred-tile).  A row i = [pr, pc, 0...]; B col j
// = [-2*tr, -2*tc, 0...]; C[r] = |p_M|^2.  acc = |p|^2 - 2 p.t; d^2 = acc +
// |t|^2.  Per the 16-bit A layout, lanes 0-15 hold K=0,1 in v16h elements
// 0,1 (lanes 16-31 hold K>=8, all zero here).  C/D layout: VGPR r, lanes
// 0-15 -> row M=r, lanes 16-31 -> M=r+8.
// ---------------------------------------------------------------------------
__global__ void habl_hausdorff_kernel(const F2* __restrict__ pts,
                                      const int* __restrict__ counts,
                                      float* __restrict__ out) {
    const int lane = threadIdx.x & 31;
    const int wid  = threadIdx.x >> 5;
    const int u    = blockIdx.x * (blockDim.x >> 5) + wid;  // wave unit
    const int NU   = NSAMP * 2 * PTILES;
    if (u >= NU) return;

    const int tile = u % PTILES;
    const int rem  = u / PTILES;
    const int loss = rem & 1;
    const int s    = rem >> 1;
    const int plist = loss ? 2 : 0;
    const int tlist = loss ? 3 : 1;

    const int pcnt = counts[s * 4 + plist];
    const int tcnt = counts[s * 4 + tlist];
    if (pcnt == 0 || tcnt == 0) return;          // empty set -> term = 0
    const int pv = pcnt < KCAP ? pcnt : KCAP;
    const int tv = tcnt < KCAP ? tcnt : KCAP;
    if (tile * 16 >= pv) return;                 // wave-uniform: EXEC stays full

    const F2* ppts = pts + (size_t)(s * 4 + plist) * LSTRIDE;
    const F2* tpts = pts + (size_t)(s * 4 + tlist) * LSTRIDE;

    const int li   = lane & 15;
    const int iIdx = tile * 16 + li;
    F2 pp = {0.0f, 0.0f};
    if (iIdx < pv) pp = ppts[iIdx];
    const float pn_lane = pp.r * pp.r + pp.c * pp.c;

    v16h a = {};
    if (lane < 16) { a[0] = (_Float16)pp.r; a[1] = (_Float16)pp.c; }

    // fold |p|^2 per output row into the WMMA C operand (constant over jt)
    v8f cpn;
    #pragma unroll
    for (int r = 0; r < 8; ++r) {
        const float lo = __shfl(pn_lane, r, 32);
        const float hi = __shfl(pn_lane, r + 8, 32);
        cpn[r] = (lane < 16) ? lo : hi;
    }

    float rmin[8];                    // running min of d^2 (sqrt deferred)
    #pragma unroll
    for (int r = 0; r < 8; ++r) rmin[r] = __builtin_inff();

    const int ttiles = (tv + 15) >> 4;
    for (int jt = 0; jt < ttiles; ++jt) {
        const int j = jt * 16 + li;
        // sentinel coords for padded targets: d^2 ~ 7.6e8 >> 522242 max real
        F2 tp = {TSENT, TSENT};
        if (j < tv) tp = tpts[j];
        const float tn = tp.r * tp.r + tp.c * tp.c;

        v16h b = {};
        if (lane < 16) {
            b[0] = (_Float16)(-2.0f * tp.r);
            b[1] = (_Float16)(-2.0f * tp.c);
        }
        // acc[r] = |p_M|^2 - 2 p_M . t_N   (exact integers in f32)
        v8f acc = __builtin_amdgcn_wmma_f32_16x16x32_f16(
            /*neg_a=*/false, a, /*neg_b=*/false, b,
            /*c_mod=*/(short)0, cpn, /*reuse_a=*/false, /*reuse_b=*/false);

        #pragma unroll
        for (int r = 0; r < 8; ++r)
            rmin[r] = fminf(rmin[r], acc[r] + tn);
    }

    // min-reduce d^2 across the 16 lanes sharing each output row (xor
    // 1,2,4,8 never crosses the half-wave boundary)
    #pragma unroll
    for (int off = 1; off < 16; off <<= 1) {
        #pragma unroll
        for (int r = 0; r < 8; ++r)
            rmin[r] = fminf(rmin[r], __shfl_xor(rmin[r], off, 32));
    }

    if (lane == 0 || lane == 16) {
        const int mbase = tile * 16 + ((lane == 16) ? 8 : 0);
        float sum = 0.0f;
        #pragma unroll
        for (int r = 0; r < 8; ++r) {
            if (mbase + r < pv)
                sum += sqrtf(fmaxf(rmin[r], 0.0f));   // 8 sqrts per wave total
        }
        // term = sum(min)/t_cnt; mean over 64 samples; (loss1+loss2)/2 -> /128
        const float contrib = sum / ((float)tcnt * (float)(2 * NSAMP));
        atomicAdd(out, contrib);
    }
}

// ---------------------------------------------------------------------------
extern "C" void kernel_launch(void* const* d_in, const int* in_sizes, int n_in,
                              void* d_out, int out_size, void* d_ws, size_t ws_size,
                              hipStream_t stream) {
    const float4* pred   = (const float4*)d_in[0];
    const float4* target = (const float4*)d_in[1];
    float* out = (float*)d_out;

    F2*  pts    = (F2*)d_ws;                                    // 2 MB
    int* counts = (int*)((char*)d_ws +
                         (size_t)NSAMP * 4 * LSTRIDE * sizeof(F2));

    habl_zero_kernel<<<1, 1, 0, stream>>>(out);
    habl_extract_kernel<<<NSAMP, 1024, 0, stream>>>(pred, target, pts, counts);

    const int NU = NSAMP * 2 * PTILES;          // 8064 wave units
    const int wavesPerBlock = 8;                // 256 threads
    const int blocks = (NU + wavesPerBlock - 1) / wavesPerBlock;
    habl_hausdorff_kernel<<<blocks, 256, 0, stream>>>(pts, counts, out);
}
